// DynamicGraphIpaFrameDenoiser_5832565588252
// MI455X (gfx1250) — compile-verified, matrix operand order checked by reading, not run on hardware
//
#include <hip/hip_runtime.h>
#include <hip/hip_bf16.h>
#include <math.h>

// ---------------------------------------------------------------------------
// DynamicGraphIpaFrameDenoiser forward for MI455X (gfx1250, wave32, WMMA).
// All linear layers -> v_wmma_f32_16x16x32_f16 (f16 in, f32 accum), LDS-tiled
// (ds_load_b128 fragment loads, vectorized staging), fused bias+ReLU epilogue.
// ---------------------------------------------------------------------------

#define NRES 1024
#define CS   256
#define CZ   128
#define CHID 256
#define NH   8
#define PQ   8
#define PV   12
#define NLAY 4
#define KNN_K 20
#define LRK_K 40
#define EPN  (KNN_K + LRK_K)      // 60 spatial edges per node
#define ESP  (NRES * EPN)         // 61440
#define SPN  20                   // seq edges per node
#define NSE  (NRES * SPN)         // 20480
#define NIN  321                  // 256 pos + 64 time + 1 nmask
#define CATW (NH * (CHID + CZ + PV * 4))   // 3456
#define CAT_OPT   2048
#define CAT_OPTN  2336
#define CAT_OPAIR 2432

#define ATT_C1 0.03608439182435161f   // sqrt(1/3)/sqrt(256)
#define ATT_C2 0.5773502691896258f    // sqrt(1/3)
#define HW_SC  0.09622504486493764f   // sqrt(1/(3*PQ*9/2)) = sqrt(1/108)

typedef __attribute__((ext_vector_type(16))) _Float16 v16h;
typedef __attribute__((ext_vector_type(8)))  _Float16 v8h;
typedef __attribute__((ext_vector_type(4)))  _Float16 v4h;
typedef __attribute__((ext_vector_type(8)))  float    v8f;

static inline int cdiv(int a, int b) { return (a + b - 1) / b; }

static __device__ inline v16h mk16(v8h lo, v8h hi) {
  v16h r;
#pragma unroll
  for (int i = 0; i < 8; ++i) { r[i] = lo[i]; r[i + 8] = hi[i]; }
  return r;
}

// ---------------------------------------------------------------------------
// WMMA GEMM: C[n,m] = act(A[n,K] @ W[K,M] + bias[m])
// Block: 256 threads = 8 waves. Block tile 64(N) x 128(M). Waves in a 2x4
// grid; each wave owns a 32x32 output = 2x2 WMMA 16x16 tiles (4 static
// v_wmma, A frag reused over 2 col tiles, B frag over 2 row tiles).
// K stepped by 32. f32 inputs converted to f16 at LDS staging.
// LDS layouts (row stride 40 halves = 80 B, 16B-aligned rows):
//   As [64][40]  row-major  -> A frag = 2x ds_load_b128 per row subtile
//   Bst[128][40] TRANSPOSED -> B frag = 2x ds_load_b128 per col subtile
// Fragment layouts per CDNA5 ISA 7.12.2 (wave32):
//   A 16x32 f16: lane L holds row M=L%16; element K-chunks {kb..kb+7,
//                kb+16..kb+23}, kb = (L>=16)*8.  B mirrors with col=L%16.
//   C/D 16x16 f32: lane L col N=L%16; VGPR r = row r (lanes 0-15) / r+8.
// ---------------------------------------------------------------------------
__global__ __launch_bounds__(256) void gemm_wmma_kernel(
    const float* __restrict__ A, const float* __restrict__ W,
    const float* __restrict__ bias, float* __restrict__ C,
    int Nn, int K, int M, int relu)
{
  __shared__ _Float16 As[64][40];
  __shared__ _Float16 Bst[128][40];
  const int rowBase = blockIdx.y * 64;
  const int colBase = blockIdx.x * 128;
  const int tid  = threadIdx.x;
  const int lane = tid & 31;
  const int wave = tid >> 5;
  const int wr = wave >> 2, wc = wave & 3;   // 2x4 wave grid
  const bool kvec = ((K & 3) == 0);

  v8f acc00 = {0.f,0.f,0.f,0.f,0.f,0.f,0.f,0.f};
  v8f acc01 = acc00, acc10 = acc00, acc11 = acc00;

  for (int k0 = 0; k0 < K; k0 += 32) {
    const bool fast = kvec && (rowBase + 64 <= Nn) && (colBase + 128 <= M) &&
                      (k0 + 32 <= K);
    if (fast) {
      // A tile 64 rows x 32 k : 512 float4 groups, 2 iters of 256 threads
#pragma unroll
      for (int it = 0; it < 2; ++it) {
        int id = tid + it * 256;
        int r = id >> 3, g = id & 7;           // g: group of 4 k
        const float* ap = &A[(size_t)(rowBase + r) * K + (k0 + g * 4)];
        float4 f = *(const float4*)ap;
        v4h h = { (_Float16)f.x, (_Float16)f.y, (_Float16)f.z, (_Float16)f.w };
        ((v4h*)As[r])[g] = h;
      }
      // W tile 32 k x 128 cols, stored transposed: 1024 groups, 4 iters
#pragma unroll
      for (int it = 0; it < 4; ++it) {
        int id = tid + it * 256;
        int c = id & 127, g = id >> 7;         // g in 0..7
        int gc = colBase + c, gk0 = k0 + g * 4;
        float f0 = W[(size_t)(gk0 + 0) * M + gc];
        float f1 = W[(size_t)(gk0 + 1) * M + gc];
        float f2 = W[(size_t)(gk0 + 2) * M + gc];
        float f3 = W[(size_t)(gk0 + 3) * M + gc];
        v4h h = { (_Float16)f0, (_Float16)f1, (_Float16)f2, (_Float16)f3 };
        ((v4h*)Bst[c])[g] = h;
      }
      if (k0 + 64 <= K) {   // prefetch next K tile (global_prefetch_b8)
        __builtin_prefetch(&A[(size_t)(rowBase + (tid >> 2)) * K + k0 + 32], 0, 1);
        __builtin_prefetch(&W[(size_t)(k0 + 32 + (tid >> 3)) * M + colBase], 0, 1);
      }
    } else {
      for (int idx = tid; idx < 64 * 32; idx += 256) {
        int r = idx >> 5, kk = idx & 31;
        int gr = rowBase + r, gk = k0 + kk;
        float v = (gr < Nn && gk < K) ? A[(size_t)gr * K + gk] : 0.f;
        As[r][kk] = (_Float16)v;
      }
      for (int idx = tid; idx < 32 * 128; idx += 256) {
        int kk = idx >> 7, c = idx & 127;
        int gk = k0 + kk, gc = colBase + c;
        float v = (gk < K && gc < M) ? W[(size_t)gk * M + gc] : 0.f;
        Bst[c][kk] = (_Float16)v;
      }
    }
    __syncthreads();

    const int ci = lane >> 4;                   // half-wave selects K chunk
    const int r0 = wr * 32 + (lane & 15);       // row subtile 0
    const int r1 = r0 + 16;                     // row subtile 1
    const int c0 = wc * 32 + (lane & 15);       // col subtile 0
    const int c1 = c0 + 16;                     // col subtile 1
    v16h a0 = mk16(((const v8h*)As[r0])[ci],  ((const v8h*)As[r0])[ci + 2]);
    v16h a1 = mk16(((const v8h*)As[r1])[ci],  ((const v8h*)As[r1])[ci + 2]);
    v16h b0 = mk16(((const v8h*)Bst[c0])[ci], ((const v8h*)Bst[c0])[ci + 2]);
    v16h b1 = mk16(((const v8h*)Bst[c1])[ci], ((const v8h*)Bst[c1])[ci + 2]);
    acc00 = __builtin_amdgcn_wmma_f32_16x16x32_f16(false, a0, false, b0, (short)0, acc00, false, false);
    acc01 = __builtin_amdgcn_wmma_f32_16x16x32_f16(false, a0, false, b1, (short)0, acc01, false, false);
    acc10 = __builtin_amdgcn_wmma_f32_16x16x32_f16(false, a1, false, b0, (short)0, acc10, false, false);
    acc11 = __builtin_amdgcn_wmma_f32_16x16x32_f16(false, a1, false, b1, (short)0, acc11, false, false);
    __syncthreads();
  }

  // epilogue: 2x2 tiles, C/D layout: lane col = L%16, rows r + 8*(L>=16)
#pragma unroll
  for (int rt = 0; rt < 2; ++rt) {
#pragma unroll
    for (int ct = 0; ct < 2; ++ct) {
      const v8f& acc = (rt == 0) ? (ct == 0 ? acc00 : acc01)
                                 : (ct == 0 ? acc10 : acc11);
      int col  = colBase + wc * 32 + ct * 16 + (lane & 15);
      int row0 = rowBase + wr * 32 + rt * 16 + ((lane >> 4) << 3);
      if (col < M) {
        float bb = bias ? bias[col] : 0.f;
#pragma unroll
        for (int i = 0; i < 8; ++i) {
          int row = row0 + i;
          if (row < Nn) {
            float v = acc[i] + bb;
            if (relu) v = fmaxf(v, 0.f);
            C[(size_t)row * M + col] = v;
          }
        }
      }
    }
  }
}

// ---------------------------------------------------------------------------
// LayerNorm (optionally fused residual u*umask on input and omask on output).
// ---------------------------------------------------------------------------
__global__ __launch_bounds__(256) void add_ln_kernel(
    const float* __restrict__ x, const float* __restrict__ u,
    const int* __restrict__ umask, const float* __restrict__ g,
    const float* __restrict__ b, float* __restrict__ y, int w,
    const int* __restrict__ omask)
{
  int row = blockIdx.x, j = threadIdx.x;
  __shared__ float red[256];
  __shared__ float sh_mean, sh_rstd;
  float val = 0.f;
  if (j < w) {
    val = x[(size_t)row * w + j];
    if (u) {
      float uu = u[(size_t)row * w + j];
      if (umask) uu *= (float)umask[row];
      val += uu;
    }
  }
  red[j] = (j < w) ? val : 0.f;
  __syncthreads();
  for (int s = 128; s > 0; s >>= 1) { if (j < s) red[j] += red[j + s]; __syncthreads(); }
  if (j == 0) sh_mean = red[0] / (float)w;
  __syncthreads();
  float mean = sh_mean;
  float dv = (j < w) ? (val - mean) : 0.f;
  red[j] = dv * dv;
  __syncthreads();
  for (int s = 128; s > 0; s >>= 1) { if (j < s) red[j] += red[j + s]; __syncthreads(); }
  if (j == 0) sh_rstd = rsqrtf(red[0] / (float)w + 1e-5f);
  __syncthreads();
  if (j < w) {
    float o = (val - mean) * sh_rstd * g[j] + b[j];
    if (omask) o *= (float)omask[row];
    y[(size_t)row * w + j] = o;
  }
}

// ------------------------- small elementwise kernels -----------------------
__global__ void fill_zero_kernel(float* p, int n) {
  int i = blockIdx.x * 256 + threadIdx.x;
  if (i < n) p[i] = 0.f;
}
__global__ void scale_kernel(float* p, float s, int n) {
  int i = blockIdx.x * 256 + threadIdx.x;
  if (i < n) p[i] *= s;
}
__global__ void copy_kernel(const float* x, float* y, int n) {
  int i = blockIdx.x * 256 + threadIdx.x;
  if (i < n) y[i] = x[i];
}
__global__ void rowmask_kernel(const float* x, const int* mask, float* y, int n, int w) {
  int i = blockIdx.x * 256 + threadIdx.x;
  if (i < n * w) y[i] = x[i] * (float)mask[i / w];
}

// ------------------------------ rigid frames -------------------------------
__device__ inline void dev_quat2rot(float w, float x, float y, float z, float* R) {
  R[0] = 1.f - 2.f * (y * y + z * z); R[1] = 2.f * (x * y - w * z); R[2] = 2.f * (x * z + w * y);
  R[3] = 2.f * (x * y + w * z); R[4] = 1.f - 2.f * (x * x + z * z); R[5] = 2.f * (y * z - w * x);
  R[6] = 2.f * (x * z - w * y); R[7] = 2.f * (y * z + w * x); R[8] = 1.f - 2.f * (x * x + y * y);
}

__global__ void init_rigid_kernel(const float* rigids, float* rot, float* transl, int N) {
  int n = blockIdx.x * 256 + threadIdx.x;
  if (n >= N) return;
  const float* q = rigids + (size_t)n * 7;
  float inv = rsqrtf(q[0]*q[0] + q[1]*q[1] + q[2]*q[2] + q[3]*q[3] + 1e-12f);
  dev_quat2rot(q[0]*inv, q[1]*inv, q[2]*inv, q[3]*inv, rot + (size_t)n * 9);
  transl[n * 3 + 0] = q[4];
  transl[n * 3 + 1] = q[5];
  transl[n * 3 + 2] = q[6];
}

__global__ __launch_bounds__(256) void center_reduce_kernel(
    const float* transl, const int* mask, float* csum, int N)
{
  __shared__ float sx[256], sy[256], sz[256], sm[256];
  int tid = threadIdx.x;
  float x = 0, y = 0, z = 0, m = 0;
  for (int i = blockIdx.x * 256 + tid; i < N; i += gridDim.x * 256) {
    float mm = (float)mask[i];
    x += transl[i * 3 + 0] * mm;
    y += transl[i * 3 + 1] * mm;
    z += transl[i * 3 + 2] * mm;
    m += mm;
  }
  sx[tid] = x; sy[tid] = y; sz[tid] = z; sm[tid] = m;
  __syncthreads();
  for (int s = 128; s > 0; s >>= 1) {
    if (tid < s) { sx[tid] += sx[tid+s]; sy[tid] += sy[tid+s]; sz[tid] += sz[tid+s]; sm[tid] += sm[tid+s]; }
    __syncthreads();
  }
  if (tid == 0) {
    atomicAdd(&csum[0], sx[0]); atomicAdd(&csum[1], sy[0]);
    atomicAdd(&csum[2], sz[0]); atomicAdd(&csum[3], sm[0]);
  }
}

__global__ void center_apply_kernel(float* transl, const float* csum, float* centerf, int N) {
  int n = blockIdx.x * 256 + threadIdx.x;
  if (n >= N) return;
  float inv = 1.f / csum[3];
  float cx = csum[0] * inv, cy = csum[1] * inv, cz = csum[2] * inv;
  transl[n * 3 + 0] -= cx; transl[n * 3 + 1] -= cy; transl[n * 3 + 2] -= cz;
  if (n == 0) { centerf[0] = cx; centerf[1] = cy; centerf[2] = cz; }
}

// ------------------------------ features -----------------------------------
__global__ void node_input_kernel(const int* residx, const int* nmask,
                                  const float* ts, float* out, int N) {
  int idx = blockIdx.x * 256 + threadIdx.x;
  if (idx >= N * NIN) return;
  int n = idx / NIN, f = idx % NIN;
  float v;
  if (f < 256) {                      // pos_embed(residx, 256): [cos128, sin128]
    int k = (f < 128) ? f : f - 128;
    float freq = expf(-(float)(2 * k) * (logf(1e4f) / 256.f));
    float ang = (float)residx[n] * freq;
    v = (f < 128) ? cosf(ang) : sinf(ang);
  } else if (f < 320) {               // fourier_time: [sin32, cos32]
    int j = f - 256;
    int k = (j < 32) ? j : j - 32;
    float freq = expf((float)k * (logf(1000.f) / 31.f));
    float ang = ts[0] * freq;
    v = (j < 32) ? sinf(ang) : cosf(ang);
  } else {
    v = (float)nmask[n];
  }
  out[idx] = v;
}

__global__ void seq_edges_kernel(int* src, int* dst, float* valid, int N) {
  int idx = blockIdx.x * 256 + threadIdx.x;
  if (idx >= N * SPN) return;
  int n = idx / SPN, oi = idx % SPN;
  int off = (oi < 10) ? (oi - 10) : (oi - 9);   // offs = [-10..-1, 1..10]
  int nbr = n + off;
  int ok = (nbr >= 0 && nbr < N) ? 1 : 0;
  if (nbr < 0) nbr = 0;
  if (nbr > N - 1) nbr = N - 1;
  src[idx] = nbr;
  dst[idx] = n;
  valid[idx] = (float)ok;
}

__global__ void edge_feat_kernel(const float* transl, const int* residx,
                                 const int* src, const int* dst,
                                 float* out, int E) {
  int idx = blockIdx.x * 256 + threadIdx.x;
  if (idx >= E * CZ) return;
  int e = idx / CZ, f = idx % CZ;
  int s = src[e], d = dst[e];
  float dx = transl[s * 3 + 0] - transl[d * 3 + 0];
  float dy = transl[s * 3 + 1] - transl[d * 3 + 1];
  float dz = transl[s * 3 + 2] - transl[d * 3 + 2];
  float dist = sqrtf(dx * dx + dy * dy + dz * dz + 1e-8f);
  float v;
  if (f < 64) {                       // rbf(d, 64, dmax=20)
    float mu = 20.f * (float)f / 63.f;
    float sg = 20.f / 64.f;
    float t = (dist - mu) / sg;
    v = expf(-t * t);
  } else {                            // pos_embed(off, 64): [cos32, sin32]
    int j = f - 64;
    int k = (j < 32) ? j : j - 32;
    float freq = expf(-(float)(2 * k) * (logf(1e4f) / 64.f));
    float ang = (float)(residx[s] - residx[d]) * freq;
    v = (j < 32) ? cosf(ang) : sinf(ang);
  }
  out[idx] = v;
}

// --------------------- spatial edge builder (KNN + gumbel LR) --------------
__device__ inline float hash01(unsigned a) {
  a ^= a >> 16; a *= 0x7feb352dU; a ^= a >> 15; a *= 0x846ca68bU; a ^= a >> 16;
  return ((float)(a >> 8) + 0.5f) * (1.f / 16777216.f);
}

__global__ __launch_bounds__(256) void build_edges_kernel(
    const float* transl, const int* mask, int* src, int layer, int N)
{
  __shared__ float dist[NRES];
  __shared__ float score[NRES];
  __shared__ float red[256];
  __shared__ int   redi[256];
  int i = blockIdx.x, tid = threadIdx.x;
  float xi = transl[i * 3 + 0], yi = transl[i * 3 + 1], zi = transl[i * 3 + 2];
  for (int j = tid; j < N; j += 256) {
    float dx = xi - transl[j * 3 + 0];
    float dy = yi - transl[j * 3 + 1];
    float dz = zi - transl[j * 3 + 2];
    float D = sqrtf(dx * dx + dy * dy + dz * dz + 1e-12f);
    float pen = ((j == i) || (mask[j] == 0)) ? 1e9f : 0.f;
    float Dp = D + pen;
    dist[j] = Dp;
    float u = hash01((unsigned)(layer * 0x9e3779b9u) ^ (unsigned)(i * NRES + j));
    float g = -logf(-logf(u));
    score[j] = -3.f * logf(Dp) + g;
  }
  __syncthreads();
  for (int t = 0; t < EPN; ++t) {
    bool knn = (t < KNN_K);
    float best = -3.0e38f; int bi = i;
    for (int j = tid; j < N; j += 256) {
      float v = knn ? -dist[j] : score[j];
      if (v > best) { best = v; bi = j; }
    }
    red[tid] = best; redi[tid] = bi;
    __syncthreads();
    for (int s = 128; s > 0; s >>= 1) {
      if (tid < s && red[tid + s] > red[tid]) { red[tid] = red[tid + s]; redi[tid] = redi[tid + s]; }
      __syncthreads();
    }
    int sel = redi[0];
    if (tid == 0) {
      src[i * EPN + t] = sel;
      if (knn) { dist[sel] = 1e30f; score[sel] -= 1e9f; }
      else     { score[sel] = -1e30f; }
    }
    __syncthreads();
  }
}

__global__ void fill_dst_kernel(int* dst, int per, int total) {
  int i = blockIdx.x * 256 + threadIdx.x;
  if (i < total) dst[i] = i / per;
}

// ------------------------------- IPA kernels -------------------------------
__global__ void rigid_apply_kernel(const float* rot, const float* transl,
                                   const float* pin, float* pout, int N, int P) {
  int idx = blockIdx.x * 256 + threadIdx.x;
  if (idx >= N * P) return;
  int n = idx / P;
  const float* R = rot + (size_t)n * 9;
  float x = pin[idx * 3], y = pin[idx * 3 + 1], z = pin[idx * 3 + 2];
  pout[idx * 3 + 0] = R[0] * x + R[1] * y + R[2] * z + transl[n * 3 + 0];
  pout[idx * 3 + 1] = R[3] * x + R[4] * y + R[5] * z + transl[n * 3 + 1];
  pout[idx * 3 + 2] = R[6] * x + R[7] * y + R[8] * z + transl[n * 3 + 2];
}

__global__ void ipa_logits_kernel(
    const float* __restrict__ q, const float* __restrict__ kv,
    const float* __restrict__ qpts, const float* __restrict__ kvpts,
    const float* __restrict__ ebias, const int* __restrict__ src,
    const int* __restrict__ dst, const float* __restrict__ evalid,
    const int* __restrict__ mask, const float* __restrict__ headw,
    float* __restrict__ a, int E)
{
  int idx = blockIdx.x * 256 + threadIdx.x;
  if (idx >= E * NH) return;
  int e = idx / NH, h = idx % NH;
  int s = src[e], d = dst[e];
  const float* qd = q + ((size_t)d * NH + h) * CHID;
  const float* ks = kv + ((size_t)s * (NH * 2) + h * 2 + 0) * CHID;
  float dot = 0.f;
  for (int c = 0; c < CHID; ++c) dot += qd[c] * ks[c];
  float val = dot * ATT_C1 + ATT_C2 * ebias[(size_t)e * NH + h];
  float d2 = 0.f;
  for (int p = 0; p < PQ; ++p) {
    for (int dd = 0; dd < 3; ++dd) {
      float qq = qpts[((size_t)d * (NH * PQ) + h * PQ + p) * 3 + dd];
      float kk = kvpts[((size_t)s * (NH * (PQ + PV)) + h * (PQ + PV) + p) * 3 + dd];
      float df = qq - kk;
      d2 += df * df;
    }
  }
  float hw = logf(1.f + expf(headw[h])) * HW_SC;   // softplus * sqrt(1/108)
  val -= 0.5f * hw * d2;
  float em = evalid ? evalid[e] : 1.f;
  val += ((float)mask[s] * (float)mask[d] * em - 1.f) * 1e5f;
  a[(size_t)e * NH + h] = val;
}

// contiguous segments: edges [n*Eper, (n+1)*Eper) all have dst == n
__global__ void seg_softmax_kernel(float* a, int N, int Eper) {
  int idx = blockIdx.x * 256 + threadIdx.x;
  if (idx >= N * NH) return;
  int n = idx / NH, h = idx % NH;
  size_t b = (size_t)n * Eper;
  float mx = -3.0e38f;
  for (int e = 0; e < Eper; ++e) mx = fmaxf(mx, a[(b + e) * NH + h]);
  float sum = 0.f;
  for (int e = 0; e < Eper; ++e) {
    float ex = expf(a[(b + e) * NH + h] - mx);
    a[(b + e) * NH + h] = ex;
    sum += ex;
  }
  float inv = 1.f / (sum + 1e-9f);
  for (int e = 0; e < Eper; ++e) a[(b + e) * NH + h] *= inv;
}

__global__ void ipa_accum_o_kernel(const float* __restrict__ w,
                                   const float* __restrict__ kv,
                                   const int* __restrict__ src,
                                   float* __restrict__ o, int N, int Eper) {
  int idx = blockIdx.x * 256 + threadIdx.x;
  if (idx >= N * NH * CHID) return;
  int c = idx % CHID, h = (idx / CHID) % NH, n = idx / (CHID * NH);
  size_t e0 = (size_t)n * Eper;
  float acc = 0.f;
  for (int e = 0; e < Eper; ++e) {
    int s = src[e0 + e];
    acc += w[(e0 + e) * NH + h] * kv[((size_t)s * (NH * 2) + h * 2 + 1) * CHID + c];
  }
  o[((size_t)n * NH + h) * CHID + c] = acc;
}

__global__ void ipa_accum_opt_kernel(const float* __restrict__ w,
                                     const float* __restrict__ kvpts,
                                     const int* __restrict__ src,
                                     float* __restrict__ optg, int N, int Eper) {
  int idx = blockIdx.x * 256 + threadIdx.x;
  if (idx >= N * NH * PV * 3) return;
  int d = idx % 3, p = (idx / 3) % PV, h = (idx / (3 * PV)) % NH, n = idx / (3 * PV * NH);
  size_t e0 = (size_t)n * Eper;
  float acc = 0.f;
  for (int e = 0; e < Eper; ++e) {
    int s = src[e0 + e];
    acc += w[(e0 + e) * NH + h] *
           kvpts[((size_t)s * (NH * (PQ + PV)) + h * (PQ + PV) + PQ + p) * 3 + d];
  }
  optg[(((size_t)n * NH + h) * PV + p) * 3 + d] = acc;
}

__global__ void ipa_accum_opair_kernel(const float* __restrict__ w,
                                       const float* __restrict__ z,
                                       float* __restrict__ opair, int N, int Eper) {
  int idx = blockIdx.x * 256 + threadIdx.x;
  if (idx >= N * NH * CZ) return;
  int zc = idx % CZ, h = (idx / CZ) % NH, n = idx / (CZ * NH);
  size_t e0 = (size_t)n * Eper;
  float acc = 0.f;
  for (int e = 0; e < Eper; ++e)
    acc += w[(e0 + e) * NH + h] * z[(e0 + e) * CZ + zc];
  opair[((size_t)n * NH + h) * CZ + zc] = acc;
}

__global__ void cat_copy_kernel(const float* x, float* cat, int N, int w, int off) {
  int idx = blockIdx.x * 256 + threadIdx.x;
  if (idx >= N * w) return;
  cat[(size_t)(idx / w) * CATW + off + (idx % w)] = x[idx];
}

__global__ void cat_pt_kernel(const float* __restrict__ optg,
                              const float* __restrict__ rot,
                              const float* __restrict__ transl,
                              float* __restrict__ cat, int N) {
  int idx = blockIdx.x * 256 + threadIdx.x;
  if (idx >= N * NH * PV) return;
  int p = idx % PV, h = (idx / PV) % NH, n = idx / (PV * NH);
  const float* R = rot + (size_t)n * 9;
  float gx = optg[idx * 3 + 0] - transl[n * 3 + 0];
  float gy = optg[idx * 3 + 1] - transl[n * 3 + 1];
  float gz = optg[idx * 3 + 2] - transl[n * 3 + 2];
  float lx = R[0] * gx + R[3] * gy + R[6] * gz;   // R^T (x - t)
  float ly = R[1] * gx + R[4] * gy + R[7] * gz;
  float lz = R[2] * gx + R[5] * gy + R[8] * gz;
  int j = h * PV + p;
  float* c = cat + (size_t)n * CATW;
  c[CAT_OPT + j * 3 + 0] = lx;
  c[CAT_OPT + j * 3 + 1] = ly;
  c[CAT_OPT + j * 3 + 2] = lz;
  c[CAT_OPTN + j] = sqrtf(lx * lx + ly * ly + lz * lz + 1e-8f);
}

// ------------------------------ rigid update -------------------------------
__global__ void compose_kernel(float* rot, float* transl, const float* upd6, int N) {
  int n = blockIdx.x * 256 + threadIdx.x;
  if (n >= N) return;
  const float* u = upd6 + (size_t)n * 6;
  float inv = rsqrtf(1.f + u[0]*u[0] + u[1]*u[1] + u[2]*u[2] + 1e-12f);
  float Ru[9];
  dev_quat2rot(inv, u[0]*inv, u[1]*inv, u[2]*inv, Ru);
  float* R = rot + (size_t)n * 9;
  float Ro[9];
#pragma unroll
  for (int i = 0; i < 9; ++i) Ro[i] = R[i];
#pragma unroll
  for (int i = 0; i < 3; ++i)
#pragma unroll
    for (int j = 0; j < 3; ++j) {
      float s = 0.f;
#pragma unroll
      for (int k = 0; k < 3; ++k) s += Ro[i * 3 + k] * Ru[k * 3 + j];
      R[i * 3 + j] = s;
    }
  float tx = Ro[0]*u[3] + Ro[1]*u[4] + Ro[2]*u[5];
  float ty = Ro[3]*u[3] + Ro[4]*u[4] + Ro[5]*u[5];
  float tz = Ro[6]*u[3] + Ro[7]*u[4] + Ro[8]*u[5];
  transl[n * 3 + 0] += tx; transl[n * 3 + 1] += ty; transl[n * 3 + 2] += tz;
}

__global__ void t7_kernel(const float* rot, const float* transl,
                          const float* centerf, float* out7, int N) {
  int n = blockIdx.x * 256 + threadIdx.x;
  if (n >= N) return;
  const float* R = rot + (size_t)n * 9;
  float w = 0.5f * sqrtf(fmaxf(1.f + R[0] + R[4] + R[8], 1e-8f));
  float inv4w = 1.f / (4.f * w);
  float* o = out7 + (size_t)n * 7;
  o[0] = w;
  o[1] = (R[7] - R[5]) * inv4w;
  o[2] = (R[2] - R[6]) * inv4w;
  o[3] = (R[3] - R[1]) * inv4w;
  o[4] = transl[n * 3 + 0] * 10.f + centerf[0];
  o[5] = transl[n * 3 + 1] * 10.f + centerf[1];
  o[6] = transl[n * 3 + 2] * 10.f + centerf[2];
}

__global__ void psi_kernel(const float* un, float* out, int N) {
  int n = blockIdx.x * 256 + threadIdx.x;
  if (n >= N) return;
  float a = un[n * 2], b = un[n * 2 + 1];
  float inv = rsqrtf(a * a + b * b + 1e-8f);
  out[n * 2] = a * inv;
  out[n * 2 + 1] = b * inv;
}

// ============================ host orchestration ===========================

struct MLPP { const float *b1, *w1, *b2, *w2, *b3, *w3, *lnb, *lng; };
struct IPAP { const float *biasb, *biasw, *headw, *kvb, *kvw, *kvpb, *kvpw,
              *outb, *outw, *qb, *qw, *qpb, *qpw; };
struct Scratch {
  float *q, *kv, *qpl, *qpts, *kvpl, *kvpts, *ebias, *attw;
  float *o, *optg, *opair, *cat;
};

struct Arena {
  char* base; size_t off;
  Arena(void* p) : base((char*)p), off(0) {}
  float* f(size_t n) { float* r = (float*)(base + off); off += ((n * sizeof(float)) + 255) & ~(size_t)255; return r; }
  int*   i(size_t n) { int*   r = (int*)  (base + off); off += ((n * sizeof(int))   + 255) & ~(size_t)255; return r; }
};

static void gemm(const float* A, const float* W, const float* bias, float* C,
                 int n, int k, int m, int relu, hipStream_t s) {
  dim3 grid(cdiv(m, 128), cdiv(n, 64));
  gemm_wmma_kernel<<<grid, 256, 0, s>>>(A, W, bias, C, n, k, m, relu);
}

static void add_ln(const float* x, const float* u, const int* umask,
                   const float* g, const float* b, float* y,
                   int rows, int w, const int* omask, hipStream_t s) {
  add_ln_kernel<<<rows, 256, 0, s>>>(x, u, umask, g, b, y, w, omask);
}

static void mlp3(const MLPP& p, const float* x, int n, int din, int d,
                 float* t1, float* t2, float* out, const int* omask, hipStream_t s) {
  gemm(x, p.w1, p.b1, t1, n, din, d, 1, s);
  gemm(t1, p.w2, p.b2, t2, n, d, d, 1, s);
  gemm(t2, p.w3, p.b3, t1, n, d, d, 0, s);
  add_ln(t1, nullptr, nullptr, p.lng, p.lnb, out, n, d, omask, s);
}

static void run_ipa(const IPAP& p, const float* node, const float* z,
                    const int* src, const int* dst, int E, int Eper,
                    const float* evalid, const float* rot, const float* transl,
                    const int* mask, Scratch& sc, float* updout, hipStream_t s) {
  gemm(node, p.qw,   p.qb,   sc.q,    NRES, CS, NH * CHID,          0, s);
  gemm(node, p.kvw,  p.kvb,  sc.kv,   NRES, CS, 2 * NH * CHID,      0, s);
  gemm(node, p.qpw,  p.qpb,  sc.qpl,  NRES, CS, NH * PQ * 3,        0, s);
  gemm(node, p.kvpw, p.kvpb, sc.kvpl, NRES, CS, NH * (PQ + PV) * 3, 0, s);
  rigid_apply_kernel<<<cdiv(NRES * NH * PQ, 256), 256, 0, s>>>(rot, transl, sc.qpl, sc.qpts, NRES, NH * PQ);
  rigid_apply_kernel<<<cdiv(NRES * NH * (PQ + PV), 256), 256, 0, s>>>(rot, transl, sc.kvpl, sc.kvpts, NRES, NH * (PQ + PV));
  gemm(z, p.biasw, p.biasb, sc.ebias, E, CZ, NH, 0, s);
  ipa_logits_kernel<<<cdiv(E * NH, 256), 256, 0, s>>>(sc.q, sc.kv, sc.qpts, sc.kvpts,
      sc.ebias, src, dst, evalid, mask, p.headw, sc.attw, E);
  seg_softmax_kernel<<<cdiv(NRES * NH, 256), 256, 0, s>>>(sc.attw, NRES, Eper);
  ipa_accum_o_kernel<<<cdiv(NRES * NH * CHID, 256), 256, 0, s>>>(sc.attw, sc.kv, src, sc.o, NRES, Eper);
  ipa_accum_opt_kernel<<<cdiv(NRES * NH * PV * 3, 256), 256, 0, s>>>(sc.attw, sc.kvpts, src, sc.optg, NRES, Eper);
  ipa_accum_opair_kernel<<<cdiv(NRES * NH * CZ, 256), 256, 0, s>>>(sc.attw, z, sc.opair, NRES, Eper);
  cat_copy_kernel<<<cdiv(NRES * NH * CHID, 256), 256, 0, s>>>(sc.o, sc.cat, NRES, NH * CHID, 0);
  cat_pt_kernel<<<cdiv(NRES * NH * PV, 256), 256, 0, s>>>(sc.optg, rot, transl, sc.cat, NRES);
  cat_copy_kernel<<<cdiv(NRES * NH * CZ, 256), 256, 0, s>>>(sc.opair, sc.cat, NRES, NH * CZ, CAT_OPAIR);
  gemm(sc.cat, p.outw, p.outb, updout, NRES, CATW, CS, 0, s);
}

extern "C" void kernel_launch(void* const* d_in, const int* in_sizes, int n_in,
                              void* d_out, int out_size, void* d_ws, size_t ws_size,
                              hipStream_t stream) {
  (void)in_sizes; (void)n_in; (void)out_size; (void)ws_size;
  const float* rigids    = (const float*)d_in[0];
  const float* ts        = (const float*)d_in[1];
  const int*   res_mask  = (const int*)d_in[2];
  const int*   noise_msk = (const int*)d_in[3];
  const int*   residx    = (const int*)d_in[4];

  // Param leaves: jax pytree flatten = alphabetical dict keys.
  // top: embed_node(8) < layers(4*46) < seq_edge_embed(8) < torsion(6)
  // mlp leaf order:  l1.b,l1.w,l2.b,l2.w,l3.b,l3.w,ln.b,ln.g
  // ipa leaf order:  bias.b,bias.w,head_w,kv.b,kv.w,kv_pts.b,kv_pts.w,
  //                  out.b,out.w,q.b,q.w,q_pts.b,q_pts.w     (13)
  // layer leaf order: attn_seq(13),attn_sp(13),bb.b,bb.w,edge_embed(8),
  //                   ln_s1.b,ln_s1.g,trans(8)                (46)
  auto PF = [&](int i) -> const float* { return (const float*)d_in[5 + i]; };
  auto mlpAt = [&](int b) {
    MLPP m{PF(b), PF(b+1), PF(b+2), PF(b+3), PF(b+4), PF(b+5), PF(b+6), PF(b+7)};
    return m;
  };
  auto ipaAt = [&](int b) {
    IPAP p{PF(b), PF(b+1), PF(b+2), PF(b+3), PF(b+4), PF(b+5), PF(b+6),
           PF(b+7), PF(b+8), PF(b+9), PF(b+10), PF(b+11), PF(b+12)};
    return p;
  };
  MLPP embedP  = mlpAt(0);
  MLPP seqEdgeP = mlpAt(192);
  const float* torL1b = PF(200); const float* torL1w = PF(201);
  const float* torL2b = PF(202); const float* torL2w = PF(203);
  const float* torLfb = PF(204); const float* torLfw = PF(205);

  hipStream_t s = stream;
  Arena a(d_ws);
  float* rot     = a.f((size_t)NRES * 9);
  float* transl  = a.f((size_t)NRES * 3);
  float* csum    = a.f(4);
  float* centerf = a.f(3);
  float* node    = a.f((size_t)NRES * CS);
  float* nodein  = a.f((size_t)NRES * NIN);
  float* tmpA    = a.f((size_t)ESP * CZ);
  float* tmpB    = a.f((size_t)ESP * CZ);
  float* seqfeat = a.f((size_t)NSE * CZ);
  float* efraw   = a.f((size_t)ESP * CZ);
  float* ef      = a.f((size_t)ESP * CZ);
  int*   seqsrc  = a.i(NSE);
  int*   seqdst  = a.i(NSE);
  float* seqval  = a.f(NSE);
  int*   spsrc   = a.i(ESP);
  int*   spdst   = a.i(ESP);
  float* upd     = a.f((size_t)NRES * CS);
  float* masked  = a.f((size_t)NRES * CS);
  float* upd6    = a.f((size_t)NRES * 6);
  Scratch sc;
  sc.q     = a.f((size_t)NRES * NH * CHID);
  sc.kv    = a.f((size_t)NRES * 2 * NH * CHID);
  sc.qpl   = a.f((size_t)NRES * NH * PQ * 3);
  sc.qpts  = a.f((size_t)NRES * NH * PQ * 3);
  sc.kvpl  = a.f((size_t)NRES * NH * (PQ + PV) * 3);
  sc.kvpts = a.f((size_t)NRES * NH * (PQ + PV) * 3);
  sc.ebias = a.f((size_t)ESP * NH);
  sc.attw  = a.f((size_t)ESP * NH);
  sc.o     = a.f((size_t)NRES * NH * CHID);
  sc.optg  = a.f((size_t)NRES * NH * PV * 3);
  sc.opair = a.f((size_t)NRES * NH * CZ);
  sc.cat   = a.f((size_t)NRES * CATW);

  // --- rigid frames, centering ---
  init_rigid_kernel<<<cdiv(NRES, 256), 256, 0, s>>>(rigids, rot, transl, NRES);
  fill_zero_kernel<<<1, 256, 0, s>>>(csum, 4);
  center_reduce_kernel<<<8, 256, 0, s>>>(transl, res_mask, csum, NRES);
  center_apply_kernel<<<cdiv(NRES, 256), 256, 0, s>>>(transl, csum, centerf, NRES);

  // --- sequence edges + features (centered, unscaled trans) ---
  seq_edges_kernel<<<cdiv(NSE, 256), 256, 0, s>>>(seqsrc, seqdst, seqval, NRES);
  edge_feat_kernel<<<cdiv(NSE * CZ, 256), 256, 0, s>>>(transl, residx, seqsrc, seqdst, efraw, NSE);
  mlp3(seqEdgeP, efraw, NSE, CZ, CZ, tmpA, tmpB, seqfeat, nullptr, s);

  // --- node embedding ---
  node_input_kernel<<<cdiv(NRES * NIN, 256), 256, 0, s>>>(residx, noise_msk, ts, nodein, NRES);
  mlp3(embedP, nodein, NRES, NIN, CS, tmpA, tmpB, node, res_mask, s);

  // --- layer-0 spatial edges (pre-scale), then trans *= 0.1 ---
  build_edges_kernel<<<NRES, 256, 0, s>>>(transl, res_mask, spsrc, 0, NRES);
  fill_dst_kernel<<<cdiv(ESP, 256), 256, 0, s>>>(spdst, EPN, ESP);
  edge_feat_kernel<<<cdiv(ESP * CZ, 256), 256, 0, s>>>(transl, residx, spsrc, spdst, efraw, ESP);
  scale_kernel<<<cdiv(NRES * 3, 256), 256, 0, s>>>(transl, 0.1f, NRES * 3);

  for (int i = 0; i < NLAY; ++i) {
    int L = 8 + i * 46;
    IPAP ipaSeq = ipaAt(L + 0);
    IPAP ipaSp  = ipaAt(L + 13);
    const float* bbb = PF(L + 26); const float* bbw = PF(L + 27);
    MLPP edgeP  = mlpAt(L + 28);
    const float* ln1b = PF(L + 36); const float* ln1g = PF(L + 37);
    MLPP transP = mlpAt(L + 38);

    if (i > 0) {
      build_edges_kernel<<<NRES, 256, 0, s>>>(transl, res_mask, spsrc, i, NRES);
      fill_dst_kernel<<<cdiv(ESP, 256), 256, 0, s>>>(spdst, EPN, ESP);
      edge_feat_kernel<<<cdiv(ESP * CZ, 256), 256, 0, s>>>(transl, residx, spsrc, spdst, efraw, ESP);
    }
    mlp3(edgeP, efraw, ESP, CZ, CZ, tmpA, tmpB, ef, nullptr, s);

    // spatial IPA + residual LN
    run_ipa(ipaSp, node, ef, spsrc, spdst, ESP, EPN, nullptr, rot, transl, res_mask, sc, upd, s);
    add_ln(node, upd, res_mask, ln1g, ln1b, node, NRES, CS, nullptr, s);

    // sequence IPA + residual LN (same ln_s1 params)
    run_ipa(ipaSeq, node, seqfeat, seqsrc, seqdst, NSE, SPN, seqval, rot, transl, res_mask, sc, upd, s);
    add_ln(node, upd, res_mask, ln1g, ln1b, node, NRES, CS, nullptr, s);

    // transition (residual MLP + LN, output masked)
    gemm(node, transP.w1, transP.b1, tmpA, NRES, CS, CS, 1, s);
    gemm(tmpA, transP.w2, transP.b2, tmpB, NRES, CS, CS, 1, s);
    gemm(tmpB, transP.w3, transP.b3, tmpA, NRES, CS, CS, 0, s);
    add_ln(node, tmpA, nullptr, transP.lng, transP.lnb, node, NRES, CS, res_mask, s);

    // backbone update
    rowmask_kernel<<<cdiv(NRES * CS, 256), 256, 0, s>>>(node, noise_msk, masked, NRES, CS);
    gemm(masked, bbw, bbb, upd6, NRES, CS, 6, 0, s);
    rowmask_kernel<<<cdiv(NRES * 6, 256), 256, 0, s>>>(upd6, noise_msk, upd6, NRES, 6);
    compose_kernel<<<cdiv(NRES, 256), 256, 0, s>>>(rot, transl, upd6, NRES);
  }

  // --- torsion head ---
  gemm(node, torL1w, torL1b, tmpA, NRES, CS, CS, 1, s);
  gemm(tmpA, torL2w, torL2b, tmpB, NRES, CS, CS, 1, s);
  gemm(tmpB, torLfw, torLfb, upd6, NRES, CS, 2, 0, s);

  // --- outputs: [node | t7 | psi] ---
  float* out = (float*)d_out;
  copy_kernel<<<cdiv(NRES * CS, 256), 256, 0, s>>>(node, out, NRES * CS);
  t7_kernel<<<cdiv(NRES, 256), 256, 0, s>>>(rot, transl, centerf, out + (size_t)NRES * CS, NRES);
  psi_kernel<<<cdiv(NRES, 256), 256, 0, s>>>(upd6, out + (size_t)NRES * CS + (size_t)NRES * 7, NRES);
}
